// CrAKN_91139206021794
// MI455X (gfx1250) — compile-verified, matrix operand order checked by reading, not run on hardware
//
#include <hip/hip_runtime.h>
#include <math.h>

#define NN   512
#define DD   64
#define HH   4
#define HDD  128
#define HHD  512   // H*HD

typedef __attribute__((ext_vector_type(16))) __bf16 v16bf;
typedef __attribute__((ext_vector_type(8)))  float  v8f;

// mish(x) = x*tanh(softplus(x)) = x*(t^2+2t)/(t^2+2t+2), t=e^x. Branchless.
__device__ __forceinline__ float mishf(float x) {
    const float t = __expf(fminf(x, 20.0f));
    const float n = t * (t + 2.0f);
    return x * (n / (n + 2.0f));
}

// WMMA 16x16x32 bf16 fragment coordinates (ISA 7.12.2):
//  element (row, k):  h = (kk>>3)&1, lane = row + 16*h,
//                     elem = (kk&7) + 8*(kk>>4),  kk = k%32, kstep = k/32
__device__ __forceinline__ int frag_h(int kk)    { return (kk >> 3) & 1; }
__device__ __forceinline__ int frag_elem(int kk) { return (kk & 7) + 8 * (kk >> 4); }

// ---------------------------------------------------------------------------
// Kernel 1: h = nf@emb_W+emb_b ; b0 = amds@bias_emb_W+b ; x = LN1(h)
// ---------------------------------------------------------------------------
__global__ __launch_bounds__(64)
void embed_kernel(const float* __restrict__ nf,   const float* __restrict__ amds,
                  const float* __restrict__ embW, const float* __restrict__ embB,
                  const float* __restrict__ bW,   const float* __restrict__ bB,
                  const float* __restrict__ g,    const float* __restrict__ b,
                  float* __restrict__ x, float* __restrict__ b0) {
    __shared__ float red[64];
    const int n = blockIdx.x, t = threadIdx.x;
    float h = embB[t];
    for (int k = 0; k < 256; ++k) h += nf[n*256 + k] * embW[k*DD + t];
    float bb = bB[t];
    for (int k = 0; k < 100; ++k) bb += amds[n*100 + k] * bW[k*DD + t];
    b0[n*DD + t] = bb;
    red[t] = h; __syncthreads();
    for (int o = 32; o > 0; o >>= 1) { if (t < o) red[t] += red[t+o]; __syncthreads(); }
    float mu = red[0] * (1.0f/64.0f); __syncthreads();
    float d = h - mu;
    red[t] = d*d; __syncthreads();
    for (int o = 32; o > 0; o >>= 1) { if (t < o) red[t] += red[t+o]; __syncthreads(); }
    float var = red[0] * (1.0f/64.0f);
    x[n*DD + t] = d * rsqrtf(var + 1e-5f) * g[t] + b[t];
}

// ---------------------------------------------------------------------------
// Kernel 2: qkv = x @ qkv_W[l] + qkv_b[l]
// ---------------------------------------------------------------------------
__global__ __launch_bounds__(256)
void qkv_kernel(const float* __restrict__ x, const float* __restrict__ W,
                const float* __restrict__ bvec, float* __restrict__ out) {
    const int col = blockIdx.x*256 + threadIdx.x;  // 0..1535
    const int n   = blockIdx.y;
    float a = bvec[col];
    #pragma unroll 8
    for (int d = 0; d < DD; ++d) a += x[n*DD + d] * W[d*1536 + col];
    out[n*1536 + col] = a;
}

// ---------------------------------------------------------------------------
// Kernel 2.5: pre-swizzle layer weights into WMMA-fragment-ordered bf16.
//  wf1: diff_W [64,512]  -> [32 tile][2 kstep][32 lane][16 elem]
//  wf2: bout_W [512,64]  -> [ 4 tile][16 kstep][32 lane][16 elem]
// ---------------------------------------------------------------------------
__global__ __launch_bounds__(256)
void wswizzle_kernel(const float* __restrict__ diffW, const float* __restrict__ boutW,
                     __bf16* __restrict__ wf1, __bf16* __restrict__ wf2) {
    const int idx = blockIdx.x*256 + threadIdx.x;  // 0..32767
    {   // diff_W element (k, n)
        const int k = idx >> 9, n = idx & 511;
        const int tile = n >> 4, bn = n & 15;
        const int s = k >> 5, kk = k & 31;
        const int lane = bn + 16*frag_h(kk);
        wf1[((tile*2 + s)*32 + lane)*16 + frag_elem(kk)] = (__bf16)diffW[idx];
    }
    {   // bout_W element (k, n)
        const int k = idx >> 6, n = idx & 63;
        const int tile = n >> 4, bn = n & 15;
        const int s = k >> 5, kk = k & 31;
        const int lane = bn + 16*frag_h(kk);
        wf2[((tile*16 + s)*32 + lane)*16 + frag_elem(kk)] = (__bf16)boutW[idx];
    }
}

// ---------------------------------------------------------------------------
// Kernel 3 (hot): fused pairwise-bias update, WMMA bf16, fragment-ordered IO.
//   tile = 16 (i,j) pairs (same i). 128 threads = 4 waves; wave w = head w.
//   be[16,512] = mish(bias @ diff_W + diff_b)   -> stays in LDS (frag order)
//   diffs[h,i,j] = ||be_head||                  -> lane shuffles
//   bias       = mish(be @ bout_W + bout_b)     -> in-place update
// ---------------------------------------------------------------------------
__global__ __launch_bounds__(128)
void bias_layer_kernel(const float* __restrict__ b0,
                       float* __restrict__ biasbuf,      // [N*N, D]
                       const __bf16* __restrict__ wf1,   // frag-ordered diff_W
                       const float* __restrict__ diffB,  // [HHD]
                       const __bf16* __restrict__ wf2,   // frag-ordered bout_W
                       const float* __restrict__ boutB,  // [D]
                       float* __restrict__ diffs,        // [H, N, N]
                       int layer0) {
    __shared__ __align__(32) __bf16 Afrag[2][32][16];    // bias tile, frag order
    __shared__ __align__(32) __bf16 Befrag[16][32][16];  // be tile,  frag order

    const int tid  = threadIdx.x;
    const int lane = tid & 31;
    const int wave = tid >> 5;          // 0..3 == head
    const int p0   = blockIdx.x * 16;   // first pair index (i*N + j0)
    const int i    = p0 / NN;
    const int j0   = p0 % NN;

    // stage bias tile -> LDS in WMMA fragment order
    for (int e = tid; e < 16*DD; e += 128) {
        const int m = e >> 6, d = e & 63;
        float v = layer0 ? (b0[(j0+m)*DD + d] - b0[i*DD + d])
                         : biasbuf[(p0+m)*DD + d];
        const int s = d >> 5, kk = d & 31;
        Afrag[s][m + 16*frag_h(kk)][frag_elem(kk)] = (__bf16)v;
    }
    __syncthreads();

    const int cg = lane >> 4;           // C/D: M = r + 8*cg, N = lane&15
    const int bn = lane & 15;

    const v16bf a0 = *(const v16bf*)Afrag[0][lane];
    const v16bf a1 = *(const v16bf*)Afrag[1][lane];
    const v16bf* w1v = (const v16bf*)wf1;
    const v16bf* w2v = (const v16bf*)wf2;

    // ---- matmul 1: be = mish(bias @ diff_W + diff_b); wave w -> head w ----
    float sq[8];
    #pragma unroll
    for (int r = 0; r < 8; ++r) sq[r] = 0.0f;

    #pragma unroll
    for (int t = 0; t < 8; ++t) {
        const int tile = wave*8 + t;
        const int n0 = tile*16;
        v8f c = {};
        c = __builtin_amdgcn_wmma_f32_16x16x32_bf16(false, a0, false,
                w1v[(tile*2 + 0)*32 + lane], (short)0, c, false, false);
        c = __builtin_amdgcn_wmma_f32_16x16x32_bf16(false, a1, false,
                w1v[(tile*2 + 1)*32 + lane], (short)0, c, false, false);
        // epilogue: + diff_b, mish, ||.||^2 partials, stash be in frag order
        const int nfull = n0 + bn;                 // column == matmul2 k index
        const float db = diffB[nfull];
        const int s2 = nfull >> 5, kk2 = nfull & 31;
        const int h2 = frag_h(kk2), e2 = frag_elem(kk2);
        #pragma unroll
        for (int r = 0; r < 8; ++r) {
            const float be = mishf(c[r] + db);
            sq[r] += be*be;
            Befrag[s2][(r + 8*cg) + 16*h2][e2] = (__bf16)be;
        }
    }

    // reduce ||.||^2 over the 16 lanes sharing one row; sqrt -> diffs[h,i,j]
    #pragma unroll
    for (int r = 0; r < 8; ++r) {
        float s = sq[r];
        s += __shfl_xor(s, 1); s += __shfl_xor(s, 2);
        s += __shfl_xor(s, 4); s += __shfl_xor(s, 8);
        if ((lane & 15) == 0) {
            const int m = r + 8*cg;
            diffs[(wave*NN + i)*NN + (j0 + m)] = sqrtf(s);
        }
    }
    __syncthreads();

    // ---- matmul 2: bias = mish(be @ bout_W + bout_b); wave w -> 16 cols ----
    {
        const int n0 = wave * 16;
        v8f c = {};
        #pragma unroll
        for (int s = 0; s < 16; ++s) {
            const v16bf a2 = *(const v16bf*)Befrag[s][lane];
            c = __builtin_amdgcn_wmma_f32_16x16x32_bf16(false, a2, false,
                    w2v[(wave*16 + s)*32 + lane], (short)0, c, false, false);
        }
        const float bb = boutB[n0 + bn];
        #pragma unroll
        for (int r = 0; r < 8; ++r)
            biasbuf[(p0 + r + 8*cg)*DD + n0 + bn] = mishf(c[r] + bb);
    }
}

// ---------------------------------------------------------------------------
// Kernel 4: softmax attention. one wave per (head, query row).
// ---------------------------------------------------------------------------
__global__ __launch_bounds__(256)
void attn_kernel(const float* __restrict__ qkv,    // [N, H, 3*HD]
                 const float* __restrict__ diffs,  // [H, N, N]
                 float* __restrict__ vals) {       // [N, H*HD]
    __shared__ float logit[8][NN];
    __shared__ float qbuf[8][HDD];
    const int lane = threadIdx.x & 31;
    const int w    = threadIdx.x >> 5;
    const int idx  = blockIdx.x * 8 + w;     // [0, H*N)
    const int h    = idx >> 9;
    const int n    = idx & (NN - 1);
    const float scale = 0.08838834764831845f; // 1/sqrt(128)

    #pragma unroll
    for (int t = 0; t < 4; ++t)
        qbuf[w][lane + 32*t] = qkv[(n*HH + h)*(3*HDD) + lane + 32*t];

    float mx = -3.4e38f;
    for (int m = lane; m < NN; m += 32) {
        const float* kp = qkv + (m*HH + h)*(3*HDD) + HDD;
        float dot = 0.0f;
        #pragma unroll 4
        for (int d = 0; d < HDD; ++d) dot += qbuf[w][d] * kp[d];
        const float lg = dot * scale + diffs[(h*NN + n)*NN + m];
        logit[w][m] = lg;
        mx = fmaxf(mx, lg);
    }
    #pragma unroll
    for (int o = 16; o > 0; o >>= 1) mx = fmaxf(mx, __shfl_xor(mx, o));
    float sum = 0.0f;
    for (int m = lane; m < NN; m += 32) {
        const float e = __expf(logit[w][m] - mx);
        logit[w][m] = e;
        sum += e;
    }
    #pragma unroll
    for (int o = 16; o > 0; o >>= 1) sum += __shfl_xor(sum, o);
    const float inv = 1.0f / sum;

    float acc[4] = {0.f, 0.f, 0.f, 0.f};
    for (int m = 0; m < NN; ++m) {
        const float p = logit[w][m];
        const float* vp = qkv + (m*HH + h)*(3*HDD) + 2*HDD;
        #pragma unroll
        for (int t = 0; t < 4; ++t) acc[t] += p * vp[lane + 32*t];
    }
    #pragma unroll
    for (int t = 0; t < 4; ++t)
        vals[n*HHD + h*HDD + lane + 32*t] = acc[t] * inv;
}

// ---------------------------------------------------------------------------
// Kernel 5: temp_x = vals @ o_W + o_b ; x = LN2(x + temp_x)
// ---------------------------------------------------------------------------
__global__ __launch_bounds__(64)
void oproj_ln_kernel(const float* __restrict__ vals, const float* __restrict__ oW,
                     const float* __restrict__ ob,   const float* __restrict__ g,
                     const float* __restrict__ b,    float* __restrict__ x) {
    __shared__ float red[64];
    const int n = blockIdx.x, t = threadIdx.x;
    float acc = ob[t];
    const float* vr = vals + n*HHD;
    #pragma unroll 4
    for (int k = 0; k < HHD; ++k) acc += vr[k] * oW[k*DD + t];
    const float y = x[n*DD + t] + acc;
    red[t] = y; __syncthreads();
    for (int o = 32; o > 0; o >>= 1) { if (t < o) red[t] += red[t+o]; __syncthreads(); }
    const float mu = red[0] * (1.0f/64.0f); __syncthreads();
    const float d = y - mu;
    red[t] = d*d; __syncthreads();
    for (int o = 32; o > 0; o >>= 1) { if (t < o) red[t] += red[t+o]; __syncthreads(); }
    const float var = red[0] * (1.0f/64.0f);
    x[n*DD + t] = d * rsqrtf(var + 1e-5f) * g[t] + b[t];
}

// ---------------------------------------------------------------------------
// Kernel 6: out = x @ out_W + out_b
// ---------------------------------------------------------------------------
__global__ __launch_bounds__(256)
void out_kernel(const float* __restrict__ x, const float* __restrict__ W,
                const float* __restrict__ bb, float* __restrict__ out) {
    const int n = blockIdx.x*256 + threadIdx.x;
    if (n < NN) {
        float a = bb[0];
        #pragma unroll 8
        for (int d = 0; d < DD; ++d) a += x[n*DD + d] * W[d];
        out[n] = a;
    }
}

// ---------------------------------------------------------------------------
extern "C" void kernel_launch(void* const* d_in, const int* in_sizes, int n_in,
                              void* d_out, int out_size, void* d_ws, size_t ws_size,
                              hipStream_t stream) {
    const float* nf    = (const float*)d_in[0];
    const float* amds  = (const float*)d_in[1];
    const float* embW  = (const float*)d_in[2];
    const float* embB  = (const float*)d_in[3];
    const float* bembW = (const float*)d_in[4];
    const float* bembB = (const float*)d_in[5];
    const float* ln1g  = (const float*)d_in[6];
    const float* ln1b  = (const float*)d_in[7];
    const float* ln2g  = (const float*)d_in[8];
    const float* ln2b  = (const float*)d_in[9];
    const float* qkvW  = (const float*)d_in[10];
    const float* qkvB  = (const float*)d_in[11];
    const float* diffW = (const float*)d_in[12];
    const float* diffB = (const float*)d_in[13];
    const float* oW    = (const float*)d_in[14];
    const float* oB    = (const float*)d_in[15];
    const float* boW   = (const float*)d_in[16];
    const float* boB   = (const float*)d_in[17];
    const float* outW  = (const float*)d_in[18];
    const float* outB  = (const float*)d_in[19];
    float* out = (float*)d_out;

    float* ws    = (float*)d_ws;
    float* x     = ws;                       // N*D
    float* b0    = x     + NN*DD;            // N*D
    float* qkvb  = b0    + NN*DD;            // N*3*HHD
    float* vals  = qkvb  + NN*3*HHD;         // N*HHD
    float* diffs = vals  + NN*HHD;           // H*N*N
    float* biasb = diffs + HH*NN*NN;         // N*N*D (67 MB, persists across layers)
    __bf16* wf1  = (__bf16*)(biasb + (size_t)NN*NN*DD);  // 32768 bf16 (64 KB)
    __bf16* wf2  = wf1 + 32768;                           // 32768 bf16 (64 KB)

    embed_kernel<<<NN, 64, 0, stream>>>(nf, amds, embW, embB, bembW, bembB,
                                        ln1g, ln1b, x, b0);
    for (int l = 0; l < 4; ++l) {
        qkv_kernel<<<dim3(6, NN), 256, 0, stream>>>(x, qkvW + (size_t)l*DD*3*HHD,
                                                    qkvB + (size_t)l*3*HHD, qkvb);
        wswizzle_kernel<<<128, 256, 0, stream>>>(diffW + (size_t)l*DD*HHD,
                                                 boW   + (size_t)l*HHD*DD, wf1, wf2);
        bias_layer_kernel<<<NN*NN/16, 128, 0, stream>>>(b0, biasb, wf1,
                                                        diffB + (size_t)l*HHD,
                                                        wf2,
                                                        boB + (size_t)l*DD,
                                                        diffs, l == 0 ? 1 : 0);
        attn_kernel<<<HH*NN/8, 256, 0, stream>>>(qkvb, diffs, vals);
        oproj_ln_kernel<<<NN, 64, 0, stream>>>(vals, oW + (size_t)l*HHD*DD,
                                               oB + (size_t)l*DD, ln2g, ln2b, x);
    }
    out_kernel<<<2, 256, 0, stream>>>(x, outW, outB, out);
}
